// TransformerLayer_18391049961925
// MI455X (gfx1250) — matrix operand order repaired and checked by reading, not compiled, not run on hardware
//
#include <hip/hip_runtime.h>
#include <hip/hip_bf16.h>
#include <math.h>

// ---------------------------------------------------------------------------
// Types for CDNA5 WMMA (gfx1250): bf16 16x16x32, f32 accumulate.
// ---------------------------------------------------------------------------
typedef __bf16 bf16_t;
typedef __bf16 v16bf __attribute__((ext_vector_type(16)));
typedef __bf16 v8bf  __attribute__((ext_vector_type(8)));
typedef float  v8f   __attribute__((ext_vector_type(8)));
typedef unsigned int u32x4 __attribute__((ext_vector_type(4)));
typedef int          i32x4 __attribute__((ext_vector_type(4)));
typedef int          i32x8 __attribute__((ext_vector_type(8)));

#ifndef __has_builtin
#define __has_builtin(x) 0
#endif
#if __has_builtin(__builtin_amdgcn_tensor_load_to_lds)
#define USE_TDM 1
#else
#define USE_TDM 0
#endif

union Frag {
    v16bf  v;
    v8bf   h[2];
    bf16_t e[16];
};

static __device__ __forceinline__ v8f wmma_bf16(const Frag& a, const Frag& b, v8f c) {
    return __builtin_amdgcn_wmma_f32_16x16x32_bf16(
        /*neg_a=*/false, a.v, /*neg_b=*/false, b.v,
        /*c_mod=*/(short)0, c, /*reuse_a=*/false, /*reuse_b=*/false);
}

// Model constants
#define CB   2
#define CL   8192
#define CD   1024
#define CH   16
#define CHD  64
#define CF   4096
#define CW   128
#define CM   (CB * CL)          // 16384 token rows

// ---------------------------------------------------------------------------
// Tensor Data Mover: 2D bf16 tile (tile_k x tile_rows, row stride in elems)
// from global memory into LDS.  D# per CDNA5 ISA ch8 (groups 0 and 1; groups
// 2/3 zero => 2D).  Tracked by TENSORcnt.
// ---------------------------------------------------------------------------
static __device__ __forceinline__ unsigned lds_addr32(const void* p) {
    // Generic shared-memory pointers carry the wave-relative LDS offset in
    // the low 32 bits.
    return (unsigned)(unsigned long long)(uintptr_t)p;
}

#if USE_TDM
static __device__ __forceinline__ void tdm_load_tile_bf16(
    unsigned lds_byte_addr, const bf16_t* gsrc,
    int tile_k_elems, int tile_rows, int row_stride_elems) {
    unsigned long long ga = (unsigned long long)(uintptr_t)gsrc;
    u32x4 g0;
    g0.x = 1u;                                      // count=1, user descriptor
    g0.y = lds_byte_addr;                           // LDS destination (bytes)
    g0.z = (unsigned)ga;                            // global_addr[31:0]
    g0.w = (unsigned)((ga >> 32) & 0x01FFFFFFu)     // global_addr[56:32]
           | (2u << 30);                            // type=2 ("image")
    i32x8 g1;
    g1[0] = 1 << 16;                                // data_size=1 (2 bytes)
    g1[1] = 0;                                      // tensor_dim0 lo16 <<16 (dim0=1<<30)
    g1[2] = 0x4000;                                 // tensor_dim0 hi16 | dim1 lo16<<16
    g1[3] = 0x4000 | (tile_k_elems << 16);          // dim1 hi16 | tile_dim0
    g1[4] = tile_rows & 0xFFFF;                     // tile_dim1 (tile_dim2=0)
    g1[5] = row_stride_elems;                       // tensor_dim0_stride[31:0]
    g1[6] = 0;                                      // stride0[47:32] | stride1 lo<<16
    g1[7] = 0;                                      // stride1 hi
    i32x4 z4 = {0, 0, 0, 0};
#if __clang_major__ >= 23
    i32x8 z8 = {0, 0, 0, 0, 0, 0, 0, 0};
    __builtin_amdgcn_tensor_load_to_lds(g0, g1, z4, z4, z8, 0);
#else
    __builtin_amdgcn_tensor_load_to_lds(g0, g1, z4, z4, 0);
#endif
}
#endif  // USE_TDM

// ---------------------------------------------------------------------------
// fp32 -> bf16 elementwise convert
// ---------------------------------------------------------------------------
__global__ void f32_to_bf16_kernel(const float* __restrict__ in,
                                   bf16_t* __restrict__ out, int n) {
    int i = blockIdx.x * 256 + threadIdx.x;
    if (i < n) out[i] = (bf16_t)in[i];
}

// ---------------------------------------------------------------------------
// RMSNorm: one block (256 threads) per row of 1024; output bf16.
// matches: x / (sqrt(mean(x^2)) + eps) * w
// ---------------------------------------------------------------------------
__global__ __launch_bounds__(256) void rmsnorm_bf16_kernel(
    const float* __restrict__ x, const float* __restrict__ w,
    bf16_t* __restrict__ out) {
    const int row  = blockIdx.x;
    const int tid  = threadIdx.x;
    const int lane = tid & 31;
    const int wv   = tid >> 5;
    const float* xr = x + (size_t)row * CD;

    float ss = 0.f;
    #pragma unroll
    for (int i = 0; i < CD / 256; ++i) {
        float v = xr[tid + i * 256];
        ss += v * v;
    }
    #pragma unroll
    for (int off = 16; off > 0; off >>= 1) ss += __shfl_down(ss, off, 32);

    __shared__ float red[8];
    if (lane == 0) red[wv] = ss;
    __syncthreads();
    if (tid == 0) {
        float t = 0.f;
        #pragma unroll
        for (int j = 0; j < 8; ++j) t += red[j];
        red[0] = 1.f / (sqrtf(t * (1.f / (float)CD)) + 1e-6f);
    }
    __syncthreads();
    const float inv = red[0];
    #pragma unroll
    for (int i = 0; i < CD / 256; ++i) {
        int c = tid + i * 256;
        out[(size_t)row * CD + c] = (bf16_t)(xr[c] * inv * w[c]);
    }
}

// ---------------------------------------------------------------------------
// Fallback tile copy (no TDM): 128x64 bf16 tile global -> LDS via VGPRs.
// ---------------------------------------------------------------------------
static __device__ __forceinline__ void copy_tile_vgpr(
    bf16_t* dst, const bf16_t* src, int ld_elems, int tid) {
    const float4* g = (const float4*)src;
    float4* l = (float4*)dst;
    #pragma unroll
    for (int i = 0; i < 4; ++i) {
        int idx = tid + i * 256;          // 1024 16B chunks
        int r = idx >> 3, c = idx & 7;
        l[idx] = g[(size_t)r * (ld_elems / 8) + c];
    }
}

// ---------------------------------------------------------------------------
// Tiled WMMA GEMM:  out[m,n] = sum_k A[m,k] * W[n,k] + bias[n] (+ resid)
//   A: [M, KD] bf16 row-major, W: [ND, KD] bf16 row-major.
//   Block tile 128x128, K-step 64, double-buffered LDS filled by the Tensor
//   Data Mover (per-wave 16-row slices, per-wave TENSORcnt + one barrier).
// MODE 0: bf16 out [M,ND]
// MODE 1: bf16 out, split heads -> [B, H, L, HD]
// MODE 2: f32 out [M,ND] with residual add
// ---------------------------------------------------------------------------
template <int KD, int ND, int MODE>
__global__ __launch_bounds__(256) void gemm_bf16_wmma(
    const bf16_t* __restrict__ A, const bf16_t* __restrict__ W,
    const float* __restrict__ bias, const float* resid, void* outp) {
    __shared__ __align__(16) bf16_t sA[2][128 * 64];
    __shared__ __align__(16) bf16_t sB[2][128 * 64];

    const int tid  = threadIdx.x;
    const int lane = tid & 31;
    const int wv   = tid >> 5;
    const int wm   = wv & 3;      // 4 waves along M (32 rows each)
    const int wn   = wv >> 2;     // 2 waves along N (64 cols each)
    const int bm   = blockIdx.y;
    const int bn   = blockIdx.x;
    const int kh   = lane >> 4;
    const int ml   = lane & 15;
    const int wvu  = __builtin_amdgcn_readfirstlane(wv);  // SGPR wave id

    v8f acc[2][4];
    #pragma unroll
    for (int i = 0; i < 2; ++i)
        #pragma unroll
        for (int j = 0; j < 4; ++j)
            #pragma unroll
            for (int e = 0; e < 8; ++e) acc[i][j][e] = 0.f;

#if USE_TDM
    // Each wave DMAs its own 16-row slice of the A and B tiles.
    auto issue = [&](int buf, int k0) {
        tdm_load_tile_bf16(lds_addr32(&sA[buf][(wvu * 16) * 64]),
                           A + (size_t)(bm * 128 + wvu * 16) * KD + k0,
                           64, 16, KD);
        tdm_load_tile_bf16(lds_addr32(&sB[buf][(wvu * 16) * 64]),
                           W + (size_t)(bn * 128 + wvu * 16) * KD + k0,
                           64, 16, KD);
    };
    issue(0, 0);
#endif

    int buf = 0;
    for (int k0 = 0; k0 < KD; k0 += 64) {
#if USE_TDM
        __builtin_amdgcn_s_wait_tensorcnt(0);  // own slices landed
        __syncthreads();                       // everyone's slices landed
        if (k0 + 64 < KD) issue(buf ^ 1, k0 + 64);  // prefetch next slab
#else
        __syncthreads();
        copy_tile_vgpr(sA[buf], A + (size_t)(bm * 128) * KD + k0, KD, tid);
        copy_tile_vgpr(sB[buf], W + (size_t)(bn * 128) * KD + k0, KD, tid);
        __syncthreads();
#endif
        const bf16_t* tileA = sA[buf];
        const bf16_t* tileB = sB[buf];
        #pragma unroll
        for (int ks = 0; ks < 64; ks += 32) {
            Frag fa[2];
            #pragma unroll
            for (int tm = 0; tm < 2; ++tm) {
                const bf16_t* p = tileA + (wm * 32 + tm * 16 + ml) * 64 + ks;
                fa[tm].h[0] = *(const v8bf*)(p + kh * 8);
                fa[tm].h[1] = *(const v8bf*)(p + 16 + kh * 8);
            }
            #pragma unroll
            for (int tn = 0; tn < 4; ++tn) {
                Frag fb;
                const bf16_t* p = tileB + (wn * 64 + tn * 16 + ml) * 64 + ks;
                fb.h[0] = *(const v8bf*)(p + kh * 8);
                fb.h[1] = *(const v8bf*)(p + 16 + kh * 8);
                #pragma unroll
                for (int tm = 0; tm < 2; ++tm)
                    acc[tm][tn] = wmma_bf16(fa[tm], fb, acc[tm][tn]);
            }
        }
        buf ^= 1;
    }

    // Epilogue.  C/D layout: VGPR r -> M = r + 8*(lane>=16), N = lane%16.
    #pragma unroll
    for (int tm = 0; tm < 2; ++tm) {
        #pragma unroll
        for (int tn = 0; tn < 4; ++tn) {
            const int n  = bn * 128 + wn * 64 + tn * 16 + ml;
            const float bi = bias[n];
            #pragma unroll
            for (int r = 0; r < 8; ++r) {
                const int m = bm * 128 + wm * 32 + tm * 16 + r + (kh << 3);
                const float val = acc[tm][tn][r] + bi;
                if constexpr (MODE == 0) {
                    ((bf16_t*)outp)[(size_t)m * ND + n] = (bf16_t)val;
                } else if constexpr (MODE == 1) {
                    const int bI = m >> 13, l = m & (CL - 1);
                    const int h = n >> 6, hd = n & 63;
                    ((bf16_t*)outp)[(((size_t)(bI * CH + h) * CL) + l) * CHD + hd] =
                        (bf16_t)val;
                } else {
                    ((float*)outp)[(size_t)m * ND + n] =
                        val + resid[(size_t)m * ND + n];
                }
            }
        }
    }
}

// ---------------------------------------------------------------------------
// Fused gate/up GEMM + SiLU:  out = silu(A@Wg^T+bg) * (A@Wu^T+bu)  (bf16)
// Same TDM double-buffered pipeline with three tiles per K-step.
// ---------------------------------------------------------------------------
template <int KD, int ND>
__global__ __launch_bounds__(256) void gemm_gateup_wmma(
    const bf16_t* __restrict__ A, const bf16_t* __restrict__ Wg,
    const bf16_t* __restrict__ Wu, const float* __restrict__ bg,
    const float* __restrict__ bu, bf16_t* __restrict__ out) {
    __shared__ __align__(16) bf16_t sA[2][128 * 64];
    __shared__ __align__(16) bf16_t sG[2][128 * 64];
    __shared__ __align__(16) bf16_t sU[2][128 * 64];

    const int tid  = threadIdx.x;
    const int lane = tid & 31;
    const int wv   = tid >> 5;
    const int wm   = wv & 3;
    const int wn   = wv >> 2;
    const int bm   = blockIdx.y;
    const int bn   = blockIdx.x;
    const int kh   = lane >> 4;
    const int ml   = lane & 15;
    const int wvu  = __builtin_amdgcn_readfirstlane(wv);

    v8f ag[2][4], au[2][4];
    #pragma unroll
    for (int i = 0; i < 2; ++i)
        #pragma unroll
        for (int j = 0; j < 4; ++j)
            #pragma unroll
            for (int e = 0; e < 8; ++e) { ag[i][j][e] = 0.f; au[i][j][e] = 0.f; }

#if USE_TDM
    auto issue = [&](int buf, int k0) {
        tdm_load_tile_bf16(lds_addr32(&sA[buf][(wvu * 16) * 64]),
                           A + (size_t)(bm * 128 + wvu * 16) * KD + k0,
                           64, 16, KD);
        tdm_load_tile_bf16(lds_addr32(&sG[buf][(wvu * 16) * 64]),
                           Wg + (size_t)(bn * 128 + wvu * 16) * KD + k0,
                           64, 16, KD);
        tdm_load_tile_bf16(lds_addr32(&sU[buf][(wvu * 16) * 64]),
                           Wu + (size_t)(bn * 128 + wvu * 16) * KD + k0,
                           64, 16, KD);
    };
    issue(0, 0);
#endif

    int buf = 0;
    for (int k0 = 0; k0 < KD; k0 += 64) {
#if USE_TDM
        __builtin_amdgcn_s_wait_tensorcnt(0);
        __syncthreads();
        if (k0 + 64 < KD) issue(buf ^ 1, k0 + 64);
#else
        __syncthreads();
        copy_tile_vgpr(sA[buf], A  + (size_t)(bm * 128) * KD + k0, KD, tid);
        copy_tile_vgpr(sG[buf], Wg + (size_t)(bn * 128) * KD + k0, KD, tid);
        copy_tile_vgpr(sU[buf], Wu + (size_t)(bn * 128) * KD + k0, KD, tid);
        __syncthreads();
#endif
        const bf16_t* tileA = sA[buf];
        const bf16_t* tileG = sG[buf];
        const bf16_t* tileU = sU[buf];
        #pragma unroll
        for (int ks = 0; ks < 64; ks += 32) {
            Frag fa[2];
            #pragma unroll
            for (int tm = 0; tm < 2; ++tm) {
                const bf16_t* p = tileA + (wm * 32 + tm * 16 + ml) * 64 + ks;
                fa[tm].h[0] = *(const v8bf*)(p + kh * 8);
                fa[tm].h[1] = *(const v8bf*)(p + 16 + kh * 8);
            }
            #pragma unroll
            for (int tn = 0; tn < 4; ++tn) {
                Frag fg, fu;
                const bf16_t* pg = tileG + (wn * 64 + tn * 16 + ml) * 64 + ks;
                const bf16_t* pu = tileU + (wn * 64 + tn * 16 + ml) * 64 + ks;
                fg.h[0] = *(const v8bf*)(pg + kh * 8);
                fg.h[1] = *(const v8bf*)(pg + 16 + kh * 8);
                fu.h[0] = *(const v8bf*)(pu + kh * 8);
                fu.h[1] = *(const v8bf*)(pu + 16 + kh * 8);
                #pragma unroll
                for (int tm = 0; tm < 2; ++tm) {
                    ag[tm][tn] = wmma_bf16(fa[tm], fg, ag[tm][tn]);
                    au[tm][tn] = wmma_bf16(fa[tm], fu, au[tm][tn]);
                }
            }
        }
        buf ^= 1;
    }

    #pragma unroll
    for (int tm = 0; tm < 2; ++tm) {
        #pragma unroll
        for (int tn = 0; tn < 4; ++tn) {
            const int n = bn * 128 + wn * 64 + tn * 16 + ml;
            const float bgv = bg[n], buv = bu[n];
            #pragma unroll
            for (int r = 0; r < 8; ++r) {
                const int m = bm * 128 + wm * 32 + tm * 16 + r + (kh << 3);
                const float g = ag[tm][tn][r] + bgv;
                const float u = au[tm][tn][r] + buv;
                const float hv = (g / (1.f + __expf(-g))) * u;  // silu(g)*u
                out[(size_t)m * ND + n] = (bf16_t)hv;
            }
        }
    }
}

// ---------------------------------------------------------------------------
// Sliding-window attention, one block per (window-block, head, batch).
// Q/K/V in [B,H,L,64] bf16.  Scores 128x256 f32 in LDS; WMMA for QK^T and PV.
// Output written as [B, L, H, 64] bf16 (== [M, D] for the O-projection).
// ---------------------------------------------------------------------------
__global__ __launch_bounds__(256) void attn_wmma_kernel(
    const bf16_t* __restrict__ Q, const bf16_t* __restrict__ K,
    const bf16_t* __restrict__ V, bf16_t* __restrict__ O) {
    __shared__ __align__(16) float  sS[128 * 256];   // scores / probs (128 KB)
    __shared__ __align__(16) bf16_t sQ[128 * 64];    // 16 KB
    __shared__ __align__(16) bf16_t sKV[256 * 64];   // 32 KB: K rows, then V^T

    const int nb   = blockIdx.x;   // window block 0..63
    const int h    = blockIdx.y;
    const int b    = blockIdx.z;
    const int tid  = threadIdx.x;
    const int lane = tid & 31;
    const int wv   = tid >> 5;
    const int kh   = lane >> 4;
    const int ml   = lane & 15;
    const size_t head_base = (size_t)(b * CH + h) * CL * CHD;

    // Load Q tile [128 x 64]
    {
        const float4* g = (const float4*)(Q + head_base + (size_t)nb * CW * CHD);
        float4* s = (float4*)sQ;
        #pragma unroll
        for (int i = 0; i < 4; ++i) s[tid + i * 256] = g[tid + i * 256];
    }
    // Load K rows [256 x 64]: prev block (zeros for nb==0) then current block.
    {
        float4* s = (float4*)sKV;
        #pragma unroll
        for (int i = 0; i < 8; ++i) {
            int idx  = tid + i * 256;        // 2048 chunks, 8 per key row
            int krow = idx >> 3;
            if (nb == 0 && krow < 128) {
                float4 z; z.x = z.y = z.z = z.w = 0.f;
                s[idx] = z;
            } else {
                const float4* g =
                    (const float4*)(K + head_base + (size_t)(nb - 1) * CW * CHD);
                s[idx] = g[idx];
            }
        }
    }
    __syncthreads();

    // ---- QK^T: M=128, N=256, K=64.  Each wave: 2 M-tiles x 8 N-tiles.
    {
        const int wm = wv & 3;     // 4 waves x 32 rows
        const int wn = wv >> 2;    // 2 waves x 128 cols
        v8f sc[2][8];
        #pragma unroll
        for (int i = 0; i < 2; ++i)
            #pragma unroll
            for (int j = 0; j < 8; ++j)
                #pragma unroll
                for (int e = 0; e < 8; ++e) sc[i][j][e] = 0.f;

        #pragma unroll
        for (int ks = 0; ks < 64; ks += 32) {
            Frag fa[2];
            #pragma unroll
            for (int tm = 0; tm < 2; ++tm) {
                const bf16_t* p = sQ + (wm * 32 + tm * 16 + ml) * 64 + ks;
                fa[tm].h[0] = *(const v8bf*)(p + kh * 8);
                fa[tm].h[1] = *(const v8bf*)(p + 16 + kh * 8);
            }
            #pragma unroll
            for (int tn = 0; tn < 8; ++tn) {
                Frag fb;
                const bf16_t* p = sKV + (wn * 128 + tn * 16 + ml) * 64 + ks;
                fb.h[0] = *(const v8bf*)(p + kh * 8);
                fb.h[1] = *(const v8bf*)(p + 16 + kh * 8);
                #pragma unroll
                for (int tm = 0; tm < 2; ++tm)
                    sc[tm][tn] = wmma_bf16(fa[tm], fb, sc[tm][tn]);
            }
        }
        // Write scaled+masked scores.
        #pragma unroll
        for (int tm = 0; tm < 2; ++tm)
            #pragma unroll
            for (int tn = 0; tn < 8; ++tn)
                #pragma unroll
                for (int r = 0; r < 8; ++r) {
                    const int m = wm * 32 + tm * 16 + r + (kh << 3);
                    const int c = wn * 128 + tn * 16 + ml;
                    const float val = sc[tm][tn][r] * 0.125f;  // 1/sqrt(64)
                    const bool valid =
                        (c < CW) ? (nb > 0 && c > m) : ((c - CW) <= m);
                    sS[m * 256 + c] = valid ? val : -1e30f;
                }
    }
    __syncthreads();

    // Load V transposed into sKV (reuse): sKV[d*256 + k] = V[k, d].
    for (int i = tid; i < 256 * 64; i += 256) {
        const int kr = i >> 6, d = i & 63;
        bf16_t val;
        if (nb == 0 && kr < 128) val = (bf16_t)0.f;
        else val = V[head_base + ((size_t)(nb - 1) * CW + kr) * CHD + d];
        sKV[d * 256 + kr] = val;
    }
    // Row-wise softmax: one thread per query row.
    if (tid < 128) {
        float* row = sS + tid * 256;
        float mx = -3.0e38f;
        for (int c = 0; c < 256; ++c) mx = fmaxf(mx, row[c]);
        float sum = 0.f;
        for (int c = 0; c < 256; ++c) {
            float e = __expf(row[c] - mx);
            row[c] = e;
            sum += e;
        }
        const float inv = 1.f / sum;
        for (int c = 0; c < 256; ++c) row[c] *= inv;
    }
    __syncthreads();

    // ---- PV: M=128, K=256, N=64.  Each wave: 2 M-tiles x 2 N-tiles.
    {
        const int wm = wv & 3;
        const int wn = wv >> 2;
        v8f o[2][2];
        #pragma unroll
        for (int i = 0; i < 2; ++i)
            #pragma unroll
            for (int j = 0; j < 2; ++j)
                #pragma unroll
                for (int e = 0; e < 8; ++e) o[i][j][e] = 0.f;

        for (int ks = 0; ks < 256; ks += 32) {
            Frag fa[2];
            #pragma unroll
            for (int tm = 0; tm < 2; ++tm) {
                const float* pr = sS + (wm * 32 + tm * 16 + ml) * 256 + ks;
                #pragma unroll
                for (int j = 0; j < 8; ++j) {
                    fa[tm].e[j]     = (bf16_t)pr[kh * 8 + j];
                    fa[tm].e[j + 8] = (bf16_t)pr[16 + kh * 8 + j];
                }
            }
            #pragma unroll
            for (int tn = 0; tn < 2; ++tn) {
                Frag fb;
                const bf16_t* p = sKV + (wn * 32 + tn * 16 + ml) * 256 + ks;
                fb.h[0] = *(const v8bf*)(p + kh * 8);
                fb.h[1] = *(const v8bf*)(p + 16 + kh * 8);
                #pragma unroll
                for (int tm = 0; tm < 2; ++tm)
                    o[tm][tn] = wmma_bf16(fa[tm], fb, o[tm][tn]);
            }
        }
        // Store to [B, L, H, 64]
        #pragma unroll
        for (int tm = 0; tm < 2; ++tm)
            #pragma unroll
            for (int tn = 0; tn < 2; ++tn)
                #pragma unroll
                for (int r = 0; r < 8; ++r) {
                    const int m = wm * 32 + tm * 16 + r + (kh << 3);
                    const int d = wn * 32 + tn * 16 + ml;
                    const int l = nb * CW + m;
                    O[(((size_t)b * CL + l) * CH + h) * CHD + d] =
                        (bf16_t)o[tm][tn][r];
                }
    }
}

// ---------------------------------------------------------------------------
// Host-side orchestration.
// Workspace (192 MB):
//   [0,32MB)    bf16 weights: Wq Wk Wv Wo (2MB ea) | Wg(8) Wu(8) Wd(8)
//   [32,64)     xn bf16 [M,D]         (reused for second RMSNorm)
//   [64,96)     Q bf16 [B,H,L,64]   \
//   [96,128)    K bf16              |  reused as FFN hidden bf16 [M,F]
//   [128,160)   V bf16              |  (128 MB) after attention is consumed
//   [160,192)   attn bf16 [M,D]     /
// d_out (f32 [M,D]) doubles as x1 residual buffer.
// ---------------------------------------------------------------------------
extern "C" void kernel_launch(void* const* d_in, const int* in_sizes, int n_in,
                              void* d_out, int out_size, void* d_ws,
                              size_t ws_size, hipStream_t stream) {
    const float* x   = (const float*)d_in[0];
    const float* Wq  = (const float*)d_in[1];
    const float* bq  = (const float*)d_in[2];
    const float* Wk  = (const float*)d_in[3];
    const float* bk  = (const float*)d_in[4];
    const float* Wv  = (const float*)d_in[5];
    const float* bv  = (const float*)d_in[6];
    const float* Wo  = (const float*)d_in[7];
    const float* bo  = (const float*)d_in[8];
    const float* Wg  = (const float*)d_in[9];
    const float* bg  = (const float*)d_in[10];
    const float* Wu  = (const float*)d_in[11];
    const float* bu  = (const float*)d_in[12];
    const float* Wd  = (const float*)d_in[13];
    const float* bd  = (const float*)d_in[14];
    const float* anw = (const float*)d_in[15];
    const float* fnw = (const float*)d_in[16];
    (void)in_sizes; (void)n_in; (void)out_size; (void)ws_size;

    char* ws = (char*)d_ws;
    const size_t MB = 1024ull * 1024ull;
    bf16_t* wq = (bf16_t*)(ws + 0 * MB);
    bf16_t* wk = (bf16_t*)(ws + 2 * MB);
    bf16_t* wv = (bf16_t*)(ws + 4 * MB);
    bf16_t* wo = (bf16_t*)(ws + 6 * MB);
    bf16_t* wg = (bf16_t*)(ws + 8 * MB);
    bf16_t* wu = (bf16_t*)(ws + 16 * MB);
    bf16_t* wd = (bf16_t*)(ws + 24 * MB);
    bf16_t* xn = (bf16_t*)(ws + 32 * MB);
    bf16_t* Qb = (bf16_t*)(ws + 64 * MB);
    bf16_t* Kb = (bf16_t*)(ws + 96 * MB);
    bf16_t* Vb = (bf16_t*)(ws + 128 * MB);
    bf16_t* Ab = (bf16_t*)(ws + 160 * MB);
    bf16_t* Hb = (bf16_t*)(ws + 64 * MB);   // reuse QKV+attn region
    float*  x1 = (float*)d_out;             // residual buffer == output buffer

    // 1) convert weights to bf16
    const int nDD = CD * CD, nFD = CF * CD;
    f32_to_bf16_kernel<<<dim3((nDD + 255) / 256), 256, 0, stream>>>(Wq, wq, nDD);
    f32_to_bf16_kernel<<<dim3((nDD + 255) / 256), 256, 0, stream>>>(Wk, wk, nDD);
    f32_to_bf16_kernel<<<dim3((nDD + 255) / 256), 256, 0, stream>>>(Wv, wv, nDD);
    f32_to_bf16_kernel<<<dim3((nDD + 255) / 256), 256, 0, stream>>>(Wo, wo, nDD);
    f32_to_bf16_kernel<<<dim3((nFD + 255) / 256), 256, 0, stream>>>(Wg, wg, nFD);
    f32_to_bf16_kernel<<<dim3((nFD + 255) / 256), 256, 0, stream>>>(Wu, wu, nFD);
    f32_to_bf16_kernel<<<dim3((nFD + 255) / 256), 256, 0, stream>>>(Wd, wd, nFD);

    // 2) attn RMSNorm
    rmsnorm_bf16_kernel<<<dim3(CM), 256, 0, stream>>>(x, anw, xn);

    // 3) Q/K/V projections -> [B,H,L,64] bf16
    gemm_bf16_wmma<CD, CD, 1><<<dim3(CD / 128, CM / 128), 256, 0, stream>>>(
        xn, wq, bq, nullptr, (void*)Qb);
    gemm_bf16_wmma<CD, CD, 1><<<dim3(CD / 128, CM / 128), 256, 0, stream>>>(
        xn, wk, bk, nullptr, (void*)Kb);
    gemm_bf16_wmma<CD, CD, 1><<<dim3(CD / 128, CM / 128), 256, 0, stream>>>(
        xn, wv, bv, nullptr, (void*)Vb);

    // 4) sliding-window attention
    attn_wmma_kernel<<<dim3(CL / CW, CH, CB), 256, 0, stream>>>(Qb, Kb, Vb, Ab);

    // 5) O projection + residual -> x1 (f32, in d_out)
    gemm_bf16_wmma<CD, CD, 2><<<dim3(CD / 128, CM / 128), 256, 0, stream>>>(
        Ab, wo, bo, x, (void*)x1);

    // 6) ffn RMSNorm (reuse xn)
    rmsnorm_bf16_kernel<<<dim3(CM), 256, 0, stream>>>(x1, fnw, xn);

    // 7) fused gate/up + SiLU -> hidden bf16 [M,F]
    gemm_gateup_wmma<CD, CF><<<dim3(CF / 128, CM / 128), 256, 0, stream>>>(
        xn, wg, wu, bg, bu, Hb);

    // 8) down projection + residual -> d_out (f32)
    gemm_bf16_wmma<CF, CD, 2><<<dim3(CD / 128, CM / 128), 256, 0, stream>>>(
        Hb, wd, bd, x1, d_out);
}
